// SLSTMNET_83288005804285
// MI455X (gfx1250) — matrix operand order repaired and checked by reading, not compiled, not run on hardware
//
#include <hip/hip_runtime.h>
#include <hip/hip_bf16.h>
#include <cstdint>

typedef __attribute__((ext_vector_type(16))) _Float16 v16h;
typedef __attribute__((ext_vector_type(4)))  _Float16 v4h;
typedef __attribute__((ext_vector_type(8)))  float    v8f;
typedef __attribute__((ext_vector_type(4)))  float    v4f;

#define B_TOT   4096
#define RSTEPS  100
#define CIN     100
#define HID     50
#define GATES   200
#define NTILES  13          // 208 padded gate columns
#define NPAD    208
#define KT_IN   4           // K = 128 (pad of 100)
#define KT_H    2           // K = 64  (pad of 50)
#define FRAG_H  512         // halves per B fragment (32 lanes * 16)

// workspace byte offsets (device scratch)
#define WS_WIH  0
#define WS_WHH  (WS_WIH + NTILES*KT_IN*FRAG_H*2)   // 53248
#define WS_FC   (WS_WHH + NTILES*KT_H*FRAG_H*2)    // 79872
#define WS_BIAS (WS_FC  + KT_H*FRAG_H*2)           // 81920  (208 floats follow)

__device__ __forceinline__ uint32_t mix32(uint32_t h) {
    h ^= h >> 16; h *= 0x7feb352du;
    h ^= h >> 15; h *= 0x846ca68bu;
    h ^= h >> 16;
    return h;
}

#if __has_builtin(__builtin_amdgcn_tanhf)
__device__ __forceinline__ float ftanh(float v) {
    return __builtin_amdgcn_tanhf(v);          // native v_tanh_f32 (TRANS)
}
#else
__device__ __forceinline__ float ftanh(float v) {
    v = fminf(fmaxf(v, -15.0f), 15.0f);
    float e = __expf(2.0f * v);
    return (e - 1.0f) / (e + 1.0f);
}
#endif

__device__ __forceinline__ float fsigmoid(float v) {
    return 0.5f + 0.5f * ftanh(0.5f * v);      // one TRANS op per sigmoid
}

// ---------------------------------------------------------------------------
// Pack weights into per-lane WMMA B-fragment layout (f16) + fused gate bias.
// B fragment layout used here (contiguous k, 32x16 K-tile): lanes 0-15 hold K=0..15 of column
// N=lane, lanes 16-31 hold K=16..31 of column N=lane-16; element j of the
// lane's v16h is k_local = j (lane<16) or 16+j (lane>=16).
// ---------------------------------------------------------------------------
__global__ void slstm_pack_kernel(const float* __restrict__ W_ih,
                                  const float* __restrict__ W_hh,
                                  const float* __restrict__ b_ih,
                                  const float* __restrict__ b_hh,
                                  const float* __restrict__ fc_w,
                                  _Float16* __restrict__ wih_pk,
                                  _Float16* __restrict__ whh_pk,
                                  _Float16* __restrict__ fc_pk,
                                  float* __restrict__ bias_sum) {
    int e = blockIdx.x * blockDim.x + threadIdx.x;

    if (e < NTILES * KT_IN * FRAG_H) {            // 26624
        int j    = e & 15;
        int lane = (e >> 4) & 31;
        int kt   = (e >> 9) & 3;
        int nt   = e >> 11;
        int n = nt * 16 + (lane & 15);
        int k = kt * 32 + ((lane < 16) ? j : 16 + j);
        float v = (n < GATES && k < CIN) ? W_ih[n * CIN + k] : 0.0f;
        wih_pk[e] = (_Float16)v;
    }
    if (e < NTILES * KT_H * FRAG_H) {             // 13312
        int j    = e & 15;
        int lane = (e >> 4) & 31;
        int kt   = (e >> 9) & 1;
        int nt   = e >> 10;
        int n = nt * 16 + (lane & 15);
        int k = kt * 32 + ((lane < 16) ? j : 16 + j);
        float v = (n < GATES && k < HID) ? W_hh[n * HID + k] : 0.0f;
        whh_pk[e] = (_Float16)v;
    }
    if (e < KT_H * FRAG_H) {                      // 1024
        int j    = e & 15;
        int lane = (e >> 4) & 31;
        int kt   = e >> 9;
        int n = lane & 15;
        int k = kt * 32 + ((lane < 16) ? j : 16 + j);
        float v = (n < 3 && k < HID) ? fc_w[n * HID + k] : 0.0f;
        fc_pk[e] = (_Float16)v;
    }
    if (e < NPAD) {
        bias_sum[e] = (e < GATES) ? (b_ih[e] + b_hh[e]) : 0.0f;
    }
}

// ---------------------------------------------------------------------------
// Persistent recurrence kernel. One wave32 owns 16 batch rows for all 100
// timesteps. Gate / hidden / fc GEMMs via v_wmma_f32_16x16x32_f16.
// ---------------------------------------------------------------------------
__global__ __launch_bounds__(64)
void slstm_step_kernel(const float* __restrict__ x,
                       const _Float16* __restrict__ wih_pk,
                       const _Float16* __restrict__ whh_pk,
                       const _Float16* __restrict__ fc_pk,
                       const float* __restrict__ bias_sum,
                       const float* __restrict__ fc_b,
                       const float* __restrict__ thr_slstm_p,
                       const float* __restrict__ beta_p,
                       const float* __restrict__ thr_lif_p,
                       float* __restrict__ out) {
    // per-wave LDS slices (2 waves / block): 43,008 B total
    __shared__ __align__(32) _Float16 sh_rateA[2][KT_IN * 32 * 16]; // rate A-tiles
    __shared__ __align__(32) _Float16 sh_m1A  [2][KT_H  * 32 * 16]; // mem1 A-tiles
    __shared__ __align__(32) _Float16 sh_spkA [2][KT_H  * 32 * 16]; // spk2 A-tiles
    __shared__ __align__(16) float    sh_gates[2][16 * NPAD];       // gate accums

    const int wave = threadIdx.x >> 5;
    const int lane = threadIdx.x & 31;
    const int tile = blockIdx.x * 2 + wave;
    const int rowbase = tile * 16;

    _Float16* rateA = sh_rateA[wave];
    _Float16* m1A   = sh_m1A[wave];
    _Float16* spkA  = sh_spkA[wave];
    float*    gates = sh_gates[wave];

    const float thr_s   = thr_slstm_p[0];
    const float beta    = beta_p[0];
    const float thr_lif = thr_lif_p[0];

    // zero-pad hidden-state A buffers (k >= 50 stays zero forever)
    for (int i = lane; i < KT_H * 32 * 16; i += 32) {
        m1A[i]  = (_Float16)0.0f;
        spkA[i] = (_Float16)0.0f;
    }

    // per-lane persistent LSTM state: 25 (row, hidden-unit) pairs
    float syn[25], mem1[25];
#pragma unroll
    for (int i = 0; i < 25; ++i) { syn[i] = 0.0f; mem1[i] = 0.0f; }

    const int nCol  = lane & 15;              // C/D fragment column
    const int mBase = (lane >= 16) ? 8 : 0;   // C/D fragment row offset

    float bias_g[NTILES];
#pragma unroll
    for (int nt = 0; nt < NTILES; ++nt)
        bias_g[nt] = bias_sum[nt * 16 + nCol];
    const float bias_fc = (nCol < 3) ? fc_b[nCol] : 0.0f;

    // per-lane x staging: iteration it covers row m=it, lane covers cols 4l..4l+3
    const int cb = lane << 2;

    v8f mem = {0.f, 0.f, 0.f, 0.f, 0.f, 0.f, 0.f, 0.f};  // leaky membrane

    for (int r = 0; r < RSTEPS; ++r) {
        if (r + 1 < RSTEPS)
            __builtin_prefetch(&x[(size_t)rowbase * (RSTEPS * CIN) + (r + 1) * CIN], 0, 0);

        // ---- stage Bernoulli rate directly into WMMA A-layout in LDS ----
        // A layout used here (contiguous k, 16x32): lanes 0-15 hold K=0..15 (j=0..15) of row
        // M=lane; lanes 16-31 hold K=16..31 of row M=lane-16.
        // Any aligned 4-run of K is contiguous in LDS.
#pragma unroll 4
        for (int m = 0; m < 16; ++m) {        // one coalesced row per iter
            v4h vals = {(_Float16)0.0f, (_Float16)0.0f,
                        (_Float16)0.0f, (_Float16)0.0f};
            if (cb < CIN) {
                v4f xv = *(const v4f*)&x[(size_t)(rowbase + m) * (RSTEPS * CIN)
                                         + r * CIN + cb];
#pragma unroll
                for (int s = 0; s < 4; ++s) {
                    uint32_t h = mix32(((uint32_t)(rowbase + m) * 2654435761u) ^
                                       ((uint32_t)r * 40503u + 0x9E3779B9u) ^
                                       ((uint32_t)(cb + s) * 2246822519u));
                    float u = (float)(h >> 8) * (1.0f / 16777216.0f);
                    vals[s] = (u < xv[s]) ? (_Float16)1.0f : (_Float16)0.0f;
                }
            }
            int kt = cb >> 5;
            int kl = cb & 31;
            int lt = m + (kl & 16);
            int j  = kl & 15;
            *(v4h*)&rateA[(kt * 32 + lt) * 16 + j] = vals;  // ds_store_b64
        }
        __syncthreads();

        // ---- gate GEMMs: [16 x 128] x [128 x 208] + [16 x 64] x [64 x 208]
        v16h ax[KT_IN];
#pragma unroll
        for (int kt = 0; kt < KT_IN; ++kt)
            ax[kt] = *(const v16h*)&rateA[(kt * 32 + lane) * 16];
        v16h am[KT_H];
#pragma unroll
        for (int kt = 0; kt < KT_H; ++kt)
            am[kt] = *(const v16h*)&m1A[(kt * 32 + lane) * 16];

#pragma unroll
        for (int nt = 0; nt < NTILES; ++nt) {
            float bs = bias_g[nt];
            v8f acc0 = {bs, bs, bs, bs, bs, bs, bs, bs};
            v8f acc1 = {0.f, 0.f, 0.f, 0.f, 0.f, 0.f, 0.f, 0.f};
            {   // even/odd K-tiles on independent accumulators (break RAW chain)
                v16h b0 = *(const v16h*)&wih_pk[((nt * KT_IN + 0) * 32 + lane) * 16];
                v16h b1 = *(const v16h*)&wih_pk[((nt * KT_IN + 1) * 32 + lane) * 16];
                v16h b2 = *(const v16h*)&wih_pk[((nt * KT_IN + 2) * 32 + lane) * 16];
                v16h b3 = *(const v16h*)&wih_pk[((nt * KT_IN + 3) * 32 + lane) * 16];
                acc0 = __builtin_amdgcn_wmma_f32_16x16x32_f16(
                    false, ax[0], false, b0, (short)0, acc0, false, false);
                acc1 = __builtin_amdgcn_wmma_f32_16x16x32_f16(
                    false, ax[1], false, b1, (short)0, acc1, false, false);
                acc0 = __builtin_amdgcn_wmma_f32_16x16x32_f16(
                    false, ax[2], false, b2, (short)0, acc0, false, false);
                acc1 = __builtin_amdgcn_wmma_f32_16x16x32_f16(
                    false, ax[3], false, b3, (short)0, acc1, false, false);
            }
            {
                v16h b0 = *(const v16h*)&whh_pk[((nt * KT_H + 0) * 32 + lane) * 16];
                v16h b1 = *(const v16h*)&whh_pk[((nt * KT_H + 1) * 32 + lane) * 16];
                acc0 = __builtin_amdgcn_wmma_f32_16x16x32_f16(
                    false, am[0], false, b0, (short)0, acc0, false, false);
                acc1 = __builtin_amdgcn_wmma_f32_16x16x32_f16(
                    false, am[1], false, b1, (short)0, acc1, false, false);
            }
            v8f acc = acc0 + acc1;
            int n = nt * 16 + nCol;
#pragma unroll
            for (int jj = 0; jj < 8; ++jj)
                gates[(jj + mBase) * NPAD + n] = acc[jj];
        }
        __syncthreads();

        // ---- elementwise SLSTM update: gate order i, f, g, o ----
#pragma unroll
        for (int i = 0; i < 25; ++i) {
            int p = lane + 32 * i;
            int m = p / HID;
            int h = p % HID;
            float gi = gates[m * NPAD + h];
            float gf = gates[m * NPAD + HID + h];
            float gg = gates[m * NPAD + 2 * HID + h];
            float go = gates[m * NPAD + 3 * HID + h];
            float si = fsigmoid(gi);
            float sf = fsigmoid(gf);
            float so = fsigmoid(go);
            syn[i]  = sf * syn[i] + si * ftanh(gg);
            mem1[i] = so * ftanh(syn[i]);
            float spk2 = (mem1[i] - thr_s > 0.0f) ? 1.0f : 0.0f;
            // scatter into WMMA A-layout for next-step hidden GEMM / fc GEMM
            int kt = h >> 5;
            int kl = h & 31;
            int lt = m + (kl & 16);
            int jj = kl & 15;
            int addr = (kt * 32 + lt) * 16 + jj;
            m1A[addr]  = (_Float16)mem1[i];
            spkA[addr] = (_Float16)spk2;
        }
        __syncthreads();

        // ---- fc GEMM: [16 x 64] x [64 x 16] (3 valid columns) ----
        v8f fcacc = {bias_fc, bias_fc, bias_fc, bias_fc,
                     bias_fc, bias_fc, bias_fc, bias_fc};
#pragma unroll
        for (int kt = 0; kt < KT_H; ++kt) {
            v16h a = *(const v16h*)&spkA[(kt * 32 + lane) * 16];
            v16h b = *(const v16h*)&fc_pk[(kt * 32 + lane) * 16];
            fcacc = __builtin_amdgcn_wmma_f32_16x16x32_f16(
                false, a, false, b, (short)0, fcacc, false, false);
        }

        // ---- Leaky membrane (subtract reset from previous mem) ----
#pragma unroll
        for (int jj = 0; jj < 8; ++jj) {
            float mprev = mem[jj];
            float reset = (mprev - thr_lif > 0.0f) ? 1.0f : 0.0f;
            float mnew  = beta * mprev + fcacc[jj] - reset * thr_lif;
            mem[jj] = mnew;
            if (nCol < 3) {
                float spk = (mnew - thr_lif > 0.0f) ? 1.0f : 0.0f;
                int m = jj + mBase;
                size_t base = (size_t)r * (B_TOT * 3) +
                              (size_t)(rowbase + m) * 3 + nCol;
                out[base] = spk;
                out[(size_t)RSTEPS * B_TOT * 3 + base] = mnew;
            }
        }
    }
}

extern "C" void kernel_launch(void* const* d_in, const int* in_sizes, int n_in,
                              void* d_out, int out_size, void* d_ws, size_t ws_size,
                              hipStream_t stream) {
    const float* x       = (const float*)d_in[0];
    const float* W_ih    = (const float*)d_in[1];
    const float* W_hh    = (const float*)d_in[2];
    const float* b_ih    = (const float*)d_in[3];
    const float* b_hh    = (const float*)d_in[4];
    const float* thr_s   = (const float*)d_in[5];
    const float* fc_w    = (const float*)d_in[6];
    const float* fc_b    = (const float*)d_in[7];
    const float* beta    = (const float*)d_in[8];
    const float* thr_lif = (const float*)d_in[9];

    char* ws = (char*)d_ws;
    _Float16* wih_pk   = (_Float16*)(ws + WS_WIH);
    _Float16* whh_pk   = (_Float16*)(ws + WS_WHH);
    _Float16* fc_pk    = (_Float16*)(ws + WS_FC);
    float*    bias_sum = (float*)(ws + WS_BIAS);

    // pack weights into WMMA B-fragment layouts (f16) + fused gate bias
    {
        int n = NTILES * KT_IN * FRAG_H;   // 26624, largest domain
        dim3 grid((n + 255) / 256), block(256);
        slstm_pack_kernel<<<grid, block, 0, stream>>>(
            W_ih, W_hh, b_ih, b_hh, fc_w, wih_pk, whh_pk, fc_pk, bias_sum);
    }

    // 256 batch tiles of 16 rows: 128 blocks x 2 waves (wave32)
    {
        dim3 grid(B_TOT / 16 / 2), block(64);
        slstm_step_kernel<<<grid, block, 0, stream>>>(
            x, wih_pk, whh_pk, fc_pk, bias_sum, fc_b,
            thr_s, beta, thr_lif, (float*)d_out);
    }
}
